// MasterNet_38182259261491
// MI455X (gfx1250) — matrix operand order repaired and checked by reading, not compiled, not run on hardware
//
#include <hip/hip_runtime.h>

// ---------------------------------------------------------------------------
// CDNA5 (gfx1250) implementation of the recurrent super-sampling pipeline.
// conv1/conv2 use V_WMMA_F32_16X16X32_F16 implicit-GEMM; sampling kernels are
// bandwidth-bound VALU; conv3+blend fused scalar epilogue.
// ---------------------------------------------------------------------------

typedef __attribute__((ext_vector_type(16))) _Float16 v16h;
typedef __attribute__((ext_vector_type(8)))  float    v8f;

union AFrag { v16h v; uint4 q[2]; };

constexpr int LH = 360, LW = 640;    // low-res H,W
constexpr int HF = 720, WF = 1280;   // hi-res H,W (factor 2)
constexpr int NB = 2;                // batch
constexpr int TT = 3;                // time steps

#define DEVFN static __device__ __forceinline__

DEVFN int   clampi(int v, int lo, int hi) { return v < lo ? lo : (v > hi ? hi : v); }
DEVFN float clamp01(float v)              { return fminf(fmaxf(v, 0.f), 1.f); }

DEVFN float lin_depth(float d) {
  const float fa = 100.f, ne = 0.1f;
  float z = ne * fa / (fa - d * (fa - ne));
  return (z - ne) / (fa - ne);
}

DEVFN void cubw(float t, float w[4]) {   // PyTorch bicubic, a = -0.75
  const float a = -0.75f;
  float t2 = t * t, t3 = t2 * t;
  w[0] = a * (t3 - 2.f * t2 + t);
  w[1] = (a + 2.f) * t3 - (a + 3.f) * t2 + 1.f;
  float s = 1.f - t;
  w[2] = (a + 2.f) * s * s * s - (a + 3.f) * s * s + 1.f;
  float u = 2.f - t;
  w[3] = a * u * u * u - 5.f * a * u * u + 8.f * a * u - 4.f * a;
}

// ---------------------------------------------------------------------------
// Weight packing (once per launch): f32 -> f16, WMMA-friendly layouts.
// w1p[oc][96]: k = tap*8 + ic (k>=72 zero padded).   w2p[tap][oc][ic].
// ---------------------------------------------------------------------------
__global__ __launch_bounds__(256) void pack_w1_kernel(const float* __restrict__ w1,
                                                      _Float16* __restrict__ w1p) {
  int i = blockIdx.x * 256 + threadIdx.x;          // over 64*96
  if (i >= 64 * 96) return;
  int oc = i / 96, k = i % 96;
  float v = 0.f;
  if (k < 72) {
    int tap = k >> 3, ic = k & 7;                  // tap = ky*3+kx
    v = w1[(size_t)(oc * 8 + ic) * 9 + tap];       // [64][8][3][3]
  }
  w1p[i] = (_Float16)v;
}

__global__ __launch_bounds__(256) void pack_w2_kernel(const float* __restrict__ w2,
                                                      _Float16* __restrict__ w2p) {
  int i = blockIdx.x * 256 + threadIdx.x;          // over 9*64*64
  if (i >= 9 * 64 * 64) return;
  int tap = i >> 12, rem = i & 4095;
  int oc = rem >> 6, ic = rem & 63;
  w2p[i] = (_Float16)w2[(size_t)(oc * 64 + ic) * 9 + tap];   // [64][64][3][3]
}

// ---------------------------------------------------------------------------
// prep: frame4 = [rgb, linearized depth]; bicubic jitter-aligned upsample
// (-> fb, f32x4 channels-last) and zero-upsample+jitter-align (-> cnn_in ch0-3, f16)
// ---------------------------------------------------------------------------
__global__ __launch_bounds__(256) void prep_kernel(
    const float* __restrict__ frames_t,   // [N,3,LH,LW]
    const float* __restrict__ depths_t,   // [N,1,LH,LW]
    const float* __restrict__ jit_t,      // [N,2] (x,y)
    float4* __restrict__ fb,              // [N, HF*WF]
    _Float16* __restrict__ cnn_in)        // [N, HF*WF, 8]
{
  const int plane = HF * WF;
  const int gid = blockIdx.x * 256 + threadIdx.x;
  const int n = gid / plane;
  const int p = gid - n * plane;
  if (n >= NB) return;
  const int Y = p / WF, X = p - Y * WF;

  const float jx = jit_t[n * 2 + 0], jy = jit_t[n * 2 + 1];
  const float gx = (2.f * X + 1.f) / WF - 1.f + 2.f * (0.5f - jx) / LW;
  const float gy = (2.f * Y + 1.f) / HF - 1.f + 2.f * (0.5f - jy) / LH;
  const float ix = ((gx + 1.f) * LW - 1.f) * 0.5f;
  const float iy = ((gy + 1.f) * LH - 1.f) * 0.5f;
  const float fx = floorf(ix), fy = floorf(iy);
  const float tx = ix - fx, ty = iy - fy;
  const int xb = (int)fx, yb = (int)fy;
  float wx[4], wy[4];
  cubw(tx, wx); cubw(ty, wy);

  const float* fr = frames_t + (size_t)n * 3 * LH * LW;
  const float* dp = depths_t + (size_t)n * LH * LW;
  const size_t HW = (size_t)LH * LW;

  float a0 = 0.f, a1 = 0.f, a2 = 0.f, a3 = 0.f;
  #pragma unroll
  for (int i = 0; i < 4; ++i) {
    const int yy = clampi(yb + i - 1, 0, LH - 1);
    const float wyi = wy[i];
    #pragma unroll
    for (int j = 0; j < 4; ++j) {
      const int xx = clampi(xb + j - 1, 0, LW - 1);
      const float w = wyi * wx[j];
      const size_t o = (size_t)yy * LW + xx;
      a0 += w * fr[o];
      a1 += w * fr[HW + o];
      a2 += w * fr[2 * HW + o];
      a3 += w * lin_depth(dp[o]);
    }
  }
  const size_t idx = (size_t)n * plane + p;
  fb[idx] = make_float4(a0, a1, a2, a3);

  // zero-upsample + integer jitter align (zero pad)
  const int jix = (int)floorf(jx * 2.f), jiy = (int)floorf(jy * 2.f);
  const int ux = X - jix, uy = Y - jiy;
  float u0 = 0.f, u1 = 0.f, u2 = 0.f, u3 = 0.f;
  if (ux >= 0 && uy >= 0 && (ux & 1) == 0 && (uy & 1) == 0) {
    const size_t o = (size_t)(uy >> 1) * LW + (ux >> 1);
    u0 = fr[o]; u1 = fr[HW + o]; u2 = fr[2 * HW + o]; u3 = lin_depth(dp[o]);
  }
  _Float16* ci = cnn_in + idx * 8;
  ci[0] = (_Float16)u0; ci[1] = (_Float16)u1; ci[2] = (_Float16)u2; ci[3] = (_Float16)u3;
}

// first step: history := frame_bicubic
__global__ __launch_bounds__(256) void init_hist_kernel(
    const float4* __restrict__ fb, float4* __restrict__ histWarp,
    _Float16* __restrict__ cnn_in)
{
  const size_t idx = (size_t)blockIdx.x * 256 + threadIdx.x;
  const float4 v = fb[idx];
  histWarp[idx] = v;
  _Float16* ci = cnn_in + idx * 8 + 4;
  ci[0] = (_Float16)v.x; ci[1] = (_Float16)v.y; ci[2] = (_Float16)v.z; ci[3] = (_Float16)v.w;
}

// later steps: bilinear-resize MV, bicubic warp history, OOB -> frame_bicubic
__global__ __launch_bounds__(256) void warp_kernel(
    const float* __restrict__ mv_t,        // [N,LH,LW,2]
    const float4* __restrict__ histPrev,   // [N, HF*WF]
    const float4* __restrict__ fb,
    float4* __restrict__ histWarp,
    _Float16* __restrict__ cnn_in)
{
  const int plane = HF * WF;
  const int gid = blockIdx.x * 256 + threadIdx.x;
  const int n = gid / plane;
  const int p = gid - n * plane;
  if (n >= NB) return;
  const int Y = p / WF, X = p - Y * WF;

  // F.interpolate bilinear, align_corners=False (half-pixel, clamp >=0, edges)
  const float cx = fmaxf((X + 0.5f) * 0.5f - 0.5f, 0.f);
  const float cy = fmaxf((Y + 0.5f) * 0.5f - 0.5f, 0.f);
  int x0i = (int)floorf(cx); const float tx = cx - (float)x0i;
  int y0i = (int)floorf(cy); const float ty = cy - (float)y0i;
  const int x1i = clampi(x0i + 1, 0, LW - 1); x0i = clampi(x0i, 0, LW - 1);
  const int y1i = clampi(y0i + 1, 0, LH - 1); y0i = clampi(y0i, 0, LH - 1);

  const float* mvn = mv_t + (size_t)n * LH * LW * 2;
  const float* mr0 = mvn + (size_t)y0i * LW * 2;
  const float* mr1 = mvn + (size_t)y1i * LW * 2;
  const float gxv = (mr0[x0i*2]   * (1.f-tx) + mr0[x1i*2]   * tx) * (1.f-ty)
                  + (mr1[x0i*2]   * (1.f-tx) + mr1[x1i*2]   * tx) * ty;
  const float gyv = (mr0[x0i*2+1] * (1.f-tx) + mr0[x1i*2+1] * tx) * (1.f-ty)
                  + (mr1[x0i*2+1] * (1.f-tx) + mr1[x1i*2+1] * tx) * ty;

  const size_t idx = (size_t)n * plane + p;
  float4 res;
  if (gxv > 1.f || gxv < -1.f || gyv > 1.f || gyv < -1.f) {
    res = fb[idx];
  } else {
    const float ix = ((gxv + 1.f) * WF - 1.f) * 0.5f;
    const float iy = ((gyv + 1.f) * HF - 1.f) * 0.5f;
    const float fx = floorf(ix), fy = floorf(iy);
    const int xb = (int)fx, yb = (int)fy;
    float wx[4], wy[4];
    cubw(ix - fx, wx); cubw(iy - fy, wy);
    const float4* hn = histPrev + (size_t)n * plane;
    float a0 = 0.f, a1 = 0.f, a2 = 0.f, a3 = 0.f;
    #pragma unroll
    for (int i = 0; i < 4; ++i) {
      const int yy = clampi(yb + i - 1, 0, HF - 1);
      #pragma unroll
      for (int j = 0; j < 4; ++j) {
        const int xx = clampi(xb + j - 1, 0, WF - 1);
        const float w = wy[i] * wx[j];
        const float4 v = hn[(size_t)yy * WF + xx];
        a0 += w * v.x; a1 += w * v.y; a2 += w * v.z; a3 += w * v.w;
      }
    }
    res = make_float4(a0, a1, a2, a3);
  }
  histWarp[idx] = res;
  _Float16* ci = cnn_in + idx * 8 + 4;
  ci[0] = (_Float16)res.x; ci[1] = (_Float16)res.y; ci[2] = (_Float16)res.z; ci[3] = (_Float16)res.w;
}

// ---------------------------------------------------------------------------
// conv1: 8 -> 64, 3x3 replicate pad, ReLU.  Implicit GEMM via WMMA f16.
// K packed tap-major: k = tap*8 + ic, padded 72 -> 96 (3 K-steps of 32).
// Wave tile: M=16 pixels (along W), N=64 out channels (4 WMMA N-tiles).
// A frag (contiguous k, same as B): lane m=L&15, K chunks at {hi*16, hi*16+8} (hi = L>=16).
// ---------------------------------------------------------------------------
__global__ __launch_bounds__(256) void conv1_kernel(
    const _Float16* __restrict__ cin,   // [N, HF*WF, 8]
    const _Float16* __restrict__ w1p,   // [64][96]
    const float* __restrict__ bias,     // [64]
    _Float16* __restrict__ out)         // [N, HF*WF, 64]
{
  __shared__ __align__(32) _Float16 wlds[64 * 96];
  {
    const uint4* s = (const uint4*)w1p;
    uint4* d = (uint4*)wlds;
    #pragma unroll
    for (int i = 0; i < 3; ++i) d[threadIdx.x + 256 * i] = s[threadIdx.x + 256 * i];
  }
  __syncthreads();

  const int lane = threadIdx.x & 31;
  const int wave = threadIdx.x >> 5;
  const int n = blockIdx.z, y = blockIdx.y;
  const int x0 = (blockIdx.x * 8 + wave) * 16;
  const int m  = lane & 15;
  const int hi = lane >> 4;
  const int k0 = hi * 16;    // A chunk base within K=32
  const int kB = hi * 16;    // B K base within K=32
  const size_t plane = (size_t)HF * WF;
  const _Float16* inN = cin + (size_t)n * plane * 8;

  v8f acc0 = {}, acc1 = {}, acc2 = {}, acc3 = {};
  #pragma unroll
  for (int s = 0; s < 3; ++s) {
    AFrag a;
    #pragma unroll
    for (int j = 0; j < 2; ++j) {
      const int kb = 32 * s + k0 + 8 * j;
      const int tap = kb >> 3;               // 8 channels per tap
      uint4 q = {0u, 0u, 0u, 0u};
      if (tap < 9) {
        const int dy = tap / 3 - 1, dx = tap % 3 - 1;
        const int yy = clampi(y + dy, 0, HF - 1);
        const int xx = clampi(x0 + m + dx, 0, WF - 1);
        q = *(const uint4*)(inN + ((size_t)yy * WF + xx) * 8);
      }
      a.q[j] = q;
    }
    const _Float16* wb = &wlds[(lane & 15) * 96 + 32 * s + kB];
    const v16h b0 = *(const v16h*)(wb);
    const v16h b1 = *(const v16h*)(wb + 16 * 96);
    const v16h b2 = *(const v16h*)(wb + 32 * 96);
    const v16h b3 = *(const v16h*)(wb + 48 * 96);
    acc0 = __builtin_amdgcn_wmma_f32_16x16x32_f16(false, a.v, false, b0, (short)0, acc0, false, false);
    acc1 = __builtin_amdgcn_wmma_f32_16x16x32_f16(false, a.v, false, b1, (short)0, acc1, false, false);
    acc2 = __builtin_amdgcn_wmma_f32_16x16x32_f16(false, a.v, false, b2, (short)0, acc2, false, false);
    acc3 = __builtin_amdgcn_wmma_f32_16x16x32_f16(false, a.v, false, b3, (short)0, acc3, false, false);
  }

  const int oc = lane & 15;
  const float bb0 = bias[oc], bb1 = bias[16 + oc], bb2 = bias[32 + oc], bb3 = bias[48 + oc];
  _Float16* outN = out + (size_t)n * plane * 64;
  const size_t rowb = ((size_t)y * WF + x0) * 64;
  #pragma unroll
  for (int v = 0; v < 8; ++v) {          // D: pixel = v + hi*8, channel = oc + 16*tile
    const int px = v + hi * 8;
    _Float16* op = outN + rowb + (size_t)px * 64 + oc;
    op[0]  = (_Float16)fmaxf(acc0[v] + bb0, 0.f);
    op[16] = (_Float16)fmaxf(acc1[v] + bb1, 0.f);
    op[32] = (_Float16)fmaxf(acc2[v] + bb2, 0.f);
    op[48] = (_Float16)fmaxf(acc3[v] + bb3, 0.f);
  }
}

// ---------------------------------------------------------------------------
// conv2: 64 -> 64, 3x3 replicate pad, ReLU.  9 taps x 2 K-steps of 32 chans.
// Full 72KB f16 weight tensor staged in LDS (WGP has 320KB).
// ---------------------------------------------------------------------------
__global__ __launch_bounds__(256) void conv2_kernel(
    const _Float16* __restrict__ ain,   // [N, HF*WF, 64]
    const _Float16* __restrict__ w2p,   // [9][64][64]
    const float* __restrict__ bias,     // [64]
    _Float16* __restrict__ out)         // [N, HF*WF, 64]
{
  __shared__ __align__(32) _Float16 wlds[9 * 64 * 64];
  {
    const uint4* s = (const uint4*)w2p;
    uint4* d = (uint4*)wlds;
    #pragma unroll
    for (int i = 0; i < 18; ++i) d[threadIdx.x + 256 * i] = s[threadIdx.x + 256 * i];
  }
  __syncthreads();

  const int lane = threadIdx.x & 31;
  const int wave = threadIdx.x >> 5;
  const int n = blockIdx.z, y = blockIdx.y;
  const int x0 = (blockIdx.x * 8 + wave) * 16;
  const int m  = lane & 15;
  const int hi = lane >> 4;
  const int k0 = hi * 16;
  const int kB = hi * 16;
  const size_t plane = (size_t)HF * WF;
  const _Float16* inN = ain + (size_t)n * plane * 64;

  v8f acc0 = {}, acc1 = {}, acc2 = {}, acc3 = {};
  for (int tap = 0; tap < 9; ++tap) {
    const int dy = tap / 3 - 1, dx = tap % 3 - 1;
    const int yy = clampi(y + dy, 0, HF - 1);
    const int xx = clampi(x0 + m + dx, 0, WF - 1);
    const _Float16* pix = inN + ((size_t)yy * WF + xx) * 64;
    #pragma unroll
    for (int cbi = 0; cbi < 2; ++cbi) {
      const int cb = cbi * 32;
      AFrag a;
      a.q[0] = *(const uint4*)(pix + cb + k0);
      a.q[1] = *(const uint4*)(pix + cb + k0 + 8);
      const _Float16* wb = &wlds[((size_t)tap * 64 + (lane & 15)) * 64 + cb + kB];
      const v16h b0 = *(const v16h*)(wb);
      const v16h b1 = *(const v16h*)(wb + 16 * 64);
      const v16h b2 = *(const v16h*)(wb + 32 * 64);
      const v16h b3 = *(const v16h*)(wb + 48 * 64);
      acc0 = __builtin_amdgcn_wmma_f32_16x16x32_f16(false, a.v, false, b0, (short)0, acc0, false, false);
      acc1 = __builtin_amdgcn_wmma_f32_16x16x32_f16(false, a.v, false, b1, (short)0, acc1, false, false);
      acc2 = __builtin_amdgcn_wmma_f32_16x16x32_f16(false, a.v, false, b2, (short)0, acc2, false, false);
      acc3 = __builtin_amdgcn_wmma_f32_16x16x32_f16(false, a.v, false, b3, (short)0, acc3, false, false);
    }
  }

  const int oc = lane & 15;
  const float bb0 = bias[oc], bb1 = bias[16 + oc], bb2 = bias[32 + oc], bb3 = bias[48 + oc];
  _Float16* outN = out + (size_t)n * plane * 64;
  const size_t rowb = ((size_t)y * WF + x0) * 64;
  #pragma unroll
  for (int v = 0; v < 8; ++v) {
    const int px = v + hi * 8;
    _Float16* op = outN + rowb + (size_t)px * 64 + oc;
    op[0]  = (_Float16)fmaxf(acc0[v] + bb0, 0.f);
    op[16] = (_Float16)fmaxf(acc1[v] + bb1, 0.f);
    op[32] = (_Float16)fmaxf(acc2[v] + bb2, 0.f);
    op[48] = (_Float16)fmaxf(acc3[v] + bb3, 0.f);
  }
}

// ---------------------------------------------------------------------------
// conv3 (64 -> 2) + alpha blend + history update + recon output. Only ~13 GF,
// VALU is fine; one thread per output pixel, weights staged in LDS.
// ---------------------------------------------------------------------------
__global__ __launch_bounds__(256) void conv3_blend_kernel(
    const _Float16* __restrict__ a2,      // [N, HF*WF, 64]
    const float* __restrict__ w3,         // [2][64][3][3]
    const float* __restrict__ b3,         // [2]
    const float4* __restrict__ histWarp,
    const float4* __restrict__ fb,
    float4* __restrict__ histPrev,        // new history (input to next step)
    float* __restrict__ outT)             // d_out + t*N*3*plane
{
  __shared__ float w3s[2 * 64 * 9];
  for (int i = threadIdx.x; i < 1152; i += 256) w3s[i] = w3[i];
  __syncthreads();

  const int X = blockIdx.x * 256 + threadIdx.x;
  const int y = blockIdx.y, n = blockIdx.z;
  const size_t plane = (size_t)HF * WF;
  const size_t p = (size_t)y * WF + X;
  const _Float16* aN = a2 + (size_t)n * plane * 64;

  float r0 = 0.f, r1 = 0.f;
  for (int tap = 0; tap < 9; ++tap) {
    const int dy = tap / 3 - 1, dx = tap % 3 - 1;
    const int yy = clampi(y + dy, 0, HF - 1);
    const int xx = clampi(X + dx, 0, WF - 1);
    const _Float16* pix = aN + ((size_t)yy * WF + xx) * 64;
    #pragma unroll 8
    for (int ic = 0; ic < 64; ++ic) {
      const float av = (float)pix[ic];
      r0 = fmaf(av, w3s[ic * 9 + tap], r0);
      r1 = fmaf(av, w3s[576 + ic * 9 + tap], r1);
    }
  }
  r0 += b3[0]; r1 += b3[1];
  const float alpha = clamp01(r0);

  const size_t idx = (size_t)n * plane + p;
  const float4 hw = histWarp[idx], fv = fb[idx];
  const float h0 = clamp01(hw.x + (fv.x - hw.x) * alpha);
  const float h1 = clamp01(hw.y + (fv.y - hw.y) * alpha);
  const float h2 = clamp01(hw.z + (fv.z - hw.z) * alpha);
  const float h3 = clamp01(hw.w + (fv.w - hw.w) * alpha + r1);
  histPrev[idx] = make_float4(h0, h1, h2, h3);
  float* o = outT + (size_t)n * 3 * plane + p;
  o[0] = h0; o[plane] = h1; o[2 * plane] = h2;
}

// ---------------------------------------------------------------------------
// Host driver: sequential time loop (history recurrence), all on `stream`.
// ---------------------------------------------------------------------------
extern "C" void kernel_launch(void* const* d_in, const int* in_sizes, int n_in,
                              void* d_out, int out_size, void* d_ws, size_t ws_size,
                              hipStream_t stream) {
  (void)in_sizes; (void)n_in; (void)out_size; (void)ws_size;

  const float* frames = (const float*)d_in[0];  // [3,2,3,360,640]
  const float* depths = (const float*)d_in[1];  // [3,2,1,360,640]
  const float* mvs    = (const float*)d_in[2];  // [3,2,360,640,2]
  const float* jit    = (const float*)d_in[3];  // [3,2,2]
  const float* w1     = (const float*)d_in[4];
  const float* b1     = (const float*)d_in[5];
  const float* w2     = (const float*)d_in[6];
  const float* b2     = (const float*)d_in[7];
  const float* w3     = (const float*)d_in[8];
  const float* b3     = (const float*)d_in[9];

  const size_t plane = (size_t)HF * WF;         // 921600
  char* ws = (char*)d_ws;
  size_t off = 0;
  auto take = [&](size_t bytes) -> void* {
    void* p = ws + off;
    off += (bytes + 255) & ~(size_t)255;
    return p;
  };
  float4*   fb       = (float4*)take(NB * plane * sizeof(float4));
  float4*   histPrev = (float4*)take(NB * plane * sizeof(float4));
  float4*   histWarp = (float4*)take(NB * plane * sizeof(float4));
  _Float16* cnn_in   = (_Float16*)take(NB * plane * 8 * sizeof(_Float16));
  _Float16* act1     = (_Float16*)take(NB * plane * 64 * sizeof(_Float16));
  _Float16* act2     = (_Float16*)take(NB * plane * 64 * sizeof(_Float16));
  _Float16* w1p      = (_Float16*)take(64 * 96 * sizeof(_Float16));
  _Float16* w2p      = (_Float16*)take(9 * 64 * 64 * sizeof(_Float16));

  pack_w1_kernel<<<24, 256, 0, stream>>>(w1, w1p);
  pack_w2_kernel<<<144, 256, 0, stream>>>(w2, w2p);

  const int nElem = (int)(NB * plane);                   // 1843200
  const dim3 gFlat(nElem / 256);                         // 7200
  const dim3 gConv(WF / 128, HF, NB);                    // wave = 16px tile, 8 waves/block
  const dim3 gPix(WF / 256, HF, NB);

  for (int t = TT - 1; t >= 0; --t) {
    const float* fr_t = frames + (size_t)t * NB * 3 * LH * LW;
    const float* dp_t = depths + (size_t)t * NB * LH * LW;
    const float* mv_t = mvs    + (size_t)t * NB * LH * LW * 2;
    const float* jt_t = jit    + (size_t)t * NB * 2;

    prep_kernel<<<gFlat, 256, 0, stream>>>(fr_t, dp_t, jt_t, fb, cnn_in);
    if (t == TT - 1)
      init_hist_kernel<<<gFlat, 256, 0, stream>>>(fb, histWarp, cnn_in);
    else
      warp_kernel<<<gFlat, 256, 0, stream>>>(mv_t, histPrev, fb, histWarp, cnn_in);

    conv1_kernel<<<gConv, 256, 0, stream>>>(cnn_in, w1p, b1, act1);
    conv2_kernel<<<gConv, 256, 0, stream>>>(act1, w2p, b2, act2);

    float* outT = (float*)d_out + (size_t)t * NB * 3 * plane;
    conv3_blend_kernel<<<gPix, 256, 0, stream>>>(act2, w3, b3, histWarp, fb,
                                                 histPrev, outT);
  }
}
